// VectorQuantizer_29291676959347
// MI455X (gfx1250) — matrix operand order repaired and checked by reading, not compiled, not run on hardware
//
#include <hip/hip_runtime.h>

// ---------------------------------------------------------------------------
// VQ nearest-code search on gfx1250 (MI455X):
//   argmin_k ( |c_k|^2 - 2<z,c_k> )  (|z|^2 dropped: monotone under argmin)
// Cross terms via v_wmma_f32_16x16x32_bf16 with bf16 hi/lo split (3 WMMAs per
// k-step) for ~f32-accurate products. Codebook staged in LDS per 128-code
// chunk as hi/lo bf16 planes + exact f32 |c_k|^2.
// Each wave owns TWO 16-row A tiles -> two independent WMMA accumulation
// chains per code tile (XDL pipelining) and 2x WMMA per DS byte.
// Code-tile loop unrolled x2 only: keeps live VGPRs < 256 (no s_set_vgpr_msb
// in the hot loop, 4 waves/SIMD instead of 3).
// ---------------------------------------------------------------------------

typedef __attribute__((ext_vector_type(16))) __bf16 v16bf;
typedef __attribute__((ext_vector_type(8)))  float  v8f;

#define D_EMBED   64
#define TILES_PER_WAVE 2
#define ROWS_PER_WAVE (16 * TILES_PER_WAVE)                 // 32
#define WAVES_PER_BLOCK 8
#define ROWS_PER_BLOCK (ROWS_PER_WAVE * WAVES_PER_BLOCK)    // 256
#define CODES_PER_CHUNK 128
#define LDS_ROW_STRIDE 72     // 64 dims padded to 72 (144B, 16B aligned, conflict-free)

static __device__ __forceinline__ unsigned short f32_to_bf16_rne(float x) {
    unsigned u = __builtin_bit_cast(unsigned, x);
    unsigned r = u + 0x7FFFu + ((u >> 16) & 1u);
    return (unsigned short)(r >> 16);
}
static __device__ __forceinline__ float bf16bits_to_f32(unsigned short h) {
    return __builtin_bit_cast(float, ((unsigned)h) << 16);
}

static __device__ __forceinline__ v16bf load_b_frag(const unsigned short* p) {
    union { v16bf v; uint4 q[2]; } u;
    u.q[0] = *reinterpret_cast<const uint4*>(p);
    u.q[1] = *reinterpret_cast<const uint4*>(p + 16);
    return u.v;
}

__global__ __launch_bounds__(256) void vq_wmma_kernel(
    const float* __restrict__ inputs,    // (n_rows, 64) f32
    const float* __restrict__ codebook,  // (K, 64) f32
    int*   __restrict__ k_out,           // (n_rows,) int32
    float* __restrict__ zq_out,          // (n_rows, 64) f32
    int n_rows, int K)
{
    __shared__ unsigned short ldsHi[CODES_PER_CHUNK * LDS_ROW_STRIDE];
    __shared__ unsigned short ldsLo[CODES_PER_CHUNK * LDS_ROW_STRIDE];
    __shared__ float          c2_lds[CODES_PER_CHUNK];

    const int tid   = threadIdx.x;
    const int wave  = tid >> 5;          // 0..7
    const int lane  = tid & 31;          // wave32
    const int m     = lane & 15;         // row-in-tile / column-in-code-tile
    const int g     = lane >> 4;         // lane half

    const int row_base = blockIdx.x * ROWS_PER_BLOCK + wave * ROWS_PER_WAVE;

    // -------------------------------------------------------------------
    // Build per-wave A fragments: 2 tiles x (16 rows x 64 dims), bf16 hi/lo.
    // ISA 16-bit A 16x32 layout: lane half g, elems 0..7 = dims[32s+8g .. +7],
    // elems 8..15 = dims[32s+16+8g .. +7].
    // -------------------------------------------------------------------
    v16bf a_hi[TILES_PER_WAVE][2], a_lo[TILES_PER_WAVE][2];
    #pragma unroll
    for (int rt = 0; rt < TILES_PER_WAVE; ++rt) {
        const int row = row_base + rt * 16 + m;
        #pragma unroll
        for (int s = 0; s < 2; ++s) {
            const float* p1 = inputs + (size_t)row * D_EMBED + 32 * s + 8 * g;
            const float* p2 = p1 + 16;
            float tmp[16];
            float4 xa = *reinterpret_cast<const float4*>(p1);
            float4 xb = *reinterpret_cast<const float4*>(p1 + 4);
            float4 xc = *reinterpret_cast<const float4*>(p2);
            float4 xd = *reinterpret_cast<const float4*>(p2 + 4);
            tmp[0]=xa.x; tmp[1]=xa.y; tmp[2]=xa.z; tmp[3]=xa.w;
            tmp[4]=xb.x; tmp[5]=xb.y; tmp[6]=xb.z; tmp[7]=xb.w;
            tmp[8]=xc.x; tmp[9]=xc.y; tmp[10]=xc.z; tmp[11]=xc.w;
            tmp[12]=xd.x; tmp[13]=xd.y; tmp[14]=xd.z; tmp[15]=xd.w;
            #pragma unroll
            for (int j = 0; j < 16; ++j) {
                unsigned short h = f32_to_bf16_rne(tmp[j]);
                float residual   = tmp[j] - bf16bits_to_f32(h);
                unsigned short l = f32_to_bf16_rne(residual);
                a_hi[rt][s][j] = __builtin_bit_cast(__bf16, h);
                a_lo[rt][s][j] = __builtin_bit_cast(__bf16, l);
            }
        }
    }

    // Running per-lane argmin state: [tile][r] tracks row (rt*16 + r + 8g).
    float min_v[TILES_PER_WAVE][8];
    int   min_k[TILES_PER_WAVE][8];
    #pragma unroll
    for (int rt = 0; rt < TILES_PER_WAVE; ++rt)
        #pragma unroll
        for (int r = 0; r < 8; ++r) { min_v[rt][r] = 3.4e38f; min_k[rt][r] = 0; }

    // -------------------------------------------------------------------
    // Chunked sweep over codebook.
    // -------------------------------------------------------------------
    for (int c0 = 0; c0 < K; c0 += CODES_PER_CHUNK) {
        __syncthreads();  // protect LDS from previous-iteration readers

        // Stage hi/lo bf16 planes: thread t covers code t/2, dim half (t&1)*32.
        {
            const int code_local = tid >> 1;
            const int dbase      = (tid & 1) * 32;
            const float* src = codebook + (size_t)(c0 + code_local) * D_EMBED + dbase;
            unsigned short* dh = ldsHi + code_local * LDS_ROW_STRIDE + dbase;
            unsigned short* dl = ldsLo + code_local * LDS_ROW_STRIDE + dbase;
            #pragma unroll 8
            for (int d = 0; d < 32; ++d) {
                float x = src[d];
                unsigned short h = f32_to_bf16_rne(x);
                unsigned short l = f32_to_bf16_rne(x - bf16bits_to_f32(h));
                dh[d] = h;
                dl[d] = l;
            }
            // Exact f32 |c_k|^2 (one thread per code).
            if (tid < CODES_PER_CHUNK) {
                const float* cr = codebook + (size_t)(c0 + tid) * D_EMBED;
                float s2 = 0.0f;
                #pragma unroll 8
                for (int d = 0; d < D_EMBED; ++d) s2 = fmaf(cr[d], cr[d], s2);
                c2_lds[tid] = s2;
            }
        }
        __syncthreads();

        // 8 code tiles of 16 per chunk; every wave sweeps all of them.
        // Unroll x2 only: full unroll pushes live VGPRs past 256 (s_set_vgpr_msb
        // toggles in the hot loop + occupancy drop).
        #pragma unroll 2
        for (int t16 = 0; t16 < 8; ++t16) {
            const int code_local = t16 * 16 + m;   // B column for this lane
            // B 32x16 layout: lane half g, elems 0..7 = k[8g..8g+7], elems 8..15 = k[16+8g..16+8g+7].
            const unsigned short* bh0 = ldsHi + code_local * LDS_ROW_STRIDE + 8 * g;
            const unsigned short* bl0 = ldsLo + code_local * LDS_ROW_STRIDE + 8 * g;

            v16bf b_hi0 = load_b_frag(bh0);
            v16bf b_hi1 = load_b_frag(bh0 + 32);
            v16bf b_lo0 = load_b_frag(bl0);
            v16bf b_lo1 = load_b_frag(bl0 + 32);

            // Two independent accumulation chains (one per A row-tile).
            v8f acc[TILES_PER_WAVE];
            #pragma unroll
            for (int rt = 0; rt < TILES_PER_WAVE; ++rt) {
                acc[rt] = (v8f){};
                // k-step 0 (dims 0..31): hi*hi + hi*lo + lo*hi
                acc[rt] = __builtin_amdgcn_wmma_f32_16x16x32_bf16(false, a_hi[rt][0], false, b_hi0, (short)0, acc[rt], false, false);
                acc[rt] = __builtin_amdgcn_wmma_f32_16x16x32_bf16(false, a_hi[rt][0], false, b_lo0, (short)0, acc[rt], false, false);
                acc[rt] = __builtin_amdgcn_wmma_f32_16x16x32_bf16(false, a_lo[rt][0], false, b_hi0, (short)0, acc[rt], false, false);
                // k-step 1 (dims 32..63)
                acc[rt] = __builtin_amdgcn_wmma_f32_16x16x32_bf16(false, a_hi[rt][1], false, b_hi1, (short)0, acc[rt], false, false);
                acc[rt] = __builtin_amdgcn_wmma_f32_16x16x32_bf16(false, a_hi[rt][1], false, b_lo1, (short)0, acc[rt], false, false);
                acc[rt] = __builtin_amdgcn_wmma_f32_16x16x32_bf16(false, a_lo[rt][1], false, b_hi1, (short)0, acc[rt], false, false);
            }

            const float c2v  = c2_lds[t16 * 16 + m];
            const int   kidx = c0 + t16 * 16 + m;
            #pragma unroll
            for (int rt = 0; rt < TILES_PER_WAVE; ++rt) {
                #pragma unroll
                for (int r = 0; r < 8; ++r) {
                    float d2 = fmaf(-2.0f, acc[rt][r], c2v);
                    if (d2 < min_v[rt][r]) { min_v[rt][r] = d2; min_k[rt][r] = kidx; }
                }
            }
        }
    }

    // -------------------------------------------------------------------
    // Cross-lane argmin per row (16-lane half), then write index + gather z_q.
    // C/D layout: VGPR r holds row (r + 8g), lanes hold columns 0..15.
    // -------------------------------------------------------------------
    #pragma unroll
    for (int rt = 0; rt < TILES_PER_WAVE; ++rt) {
        #pragma unroll
        for (int r = 0; r < 8; ++r) {
            float mv = min_v[rt][r];
            int   mk = min_k[rt][r];
            #pragma unroll
            for (int mask = 1; mask < 16; mask <<= 1) {
                float ov = __shfl_xor(mv, mask, 16);
                int   ok = __shfl_xor(mk, mask, 16);
                if (ov < mv || (ov == mv && ok < mk)) { mv = ov; mk = ok; }
            }
            const int row = row_base + rt * 16 + r + 8 * g;
            if (m == 0) k_out[row] = mk;
            // All 16 lanes hold mk: cooperatively gather the 64-float code row.
            const float* crow = codebook + (size_t)mk * D_EMBED + m * 4;
            float* orow = zq_out + (size_t)row * D_EMBED + m * 4;
            float4 v = *reinterpret_cast<const float4*>(crow);
            *reinterpret_cast<float4*>(orow) = v;
        }
    }
}

extern "C" void kernel_launch(void* const* d_in, const int* in_sizes, int n_in,
                              void* d_out, int out_size, void* d_ws, size_t ws_size,
                              hipStream_t stream) {
    const float* inputs   = (const float*)d_in[0];
    const float* codebook = (const float*)d_in[1];
    const int n_rows = in_sizes[0] / D_EMBED;   // 131072
    const int K      = in_sizes[1] / D_EMBED;   // 1024

    int*   k_out  = (int*)d_out;                 // int32 bits in first n_rows slots
    float* zq_out = (float*)d_out + n_rows;      // then (n_rows, 64) f32

    dim3 grid(n_rows / ROWS_PER_BLOCK);
    dim3 block(256);
    vq_wmma_kernel<<<grid, block, 0, stream>>>(inputs, codebook, k_out, zq_out, n_rows, K);
}
